// DualMultiCopyGenerator_49692771614993
// MI455X (gfx1250) — matrix operand order repaired and checked by reading, not compiled, not run on hardware
//
#include <hip/hip_runtime.h>
#include <hip/hip_bf16.h>

// ---------------------------------------------------------------------------
// DualMultiCopyGenerator on MI455X (gfx1250, wave32, WMMA).
// All GEMMs run on v_wmma_f32_16x16x32_bf16 (f32 accumulate), inputs converted
// once fp32->bf16.  GEMM staging uses gfx1250 async global->LDS copies
// (ASYNCcnt) with double buffering so tile k+1 loads overlap tile k WMMAs.
// fc output lands directly in d_out[:, :V]; layer-norm + gate-scale happen in
// place; scatter-copy is fused via global atomicAdd.
// ---------------------------------------------------------------------------

typedef __bf16 bf16;
typedef bf16 v16bf __attribute__((ext_vector_type(16)));
typedef float v8f  __attribute__((ext_vector_type(8)));
typedef unsigned int u32x4 __attribute__((ext_vector_type(4)));

#define B_    4
#define T_    256
#define S_    256
#define D_    512
#define V_    32000
#define H_    8
#define DH_   64
#define HID_  512
#define VEXT_ (V_ + S_ + S_)   // 32512

// ---------------- CDNA5 async global->LDS copy (ASYNCcnt) ------------------
// Per lane: LDS[lds_byte_addr] = MEM[gptr .. gptr+15].  Tracked by ASYNCcnt.
__device__ __forceinline__ void async_copy_b128(unsigned lds_byte_addr, const void* gptr) {
  asm volatile("global_load_async_to_lds_b128 %0, %1, off"
               :: "v"(lds_byte_addr), "v"(gptr) : "memory");
}
__device__ __forceinline__ void wait_async_le4() {
  asm volatile("s_wait_asynccnt 0x4" ::: "memory");
}
__device__ __forceinline__ void wait_async_0() {
  asm volatile("s_wait_asynccnt 0x0" ::: "memory");
}
// flat shared pointer -> 32-bit LDS byte address (LDS aperture = high 32 bits)
__device__ __forceinline__ unsigned lds_addr(const void* p) {
  return (unsigned)(unsigned long long)p;
}

// ---------------- WMMA fragment loaders (bf16, 16x16x32) -------------------
// A (16x32, MxK): lane l: m = l&15, hi = l>>4; elems 0..7 -> k = hi*8 + i,
//                 elems 8..15 -> k = 16 + hi*8 + (i-8).  Two 16B chunks.
// B (32x16, KxN) stored as W[N][K] row-major: lane l: n = l&15, hi = l>>4;
//                 same k order as A.  Two 16B chunks.
struct FragU { union { v16bf v; u32x4 q[2]; }; };

__device__ __forceinline__ v16bf load_frag_a(const bf16* base, int ld, int lane) {
  const int m = lane & 15, hi = lane >> 4;
  const bf16* r = base + m * ld;
  FragU f;
  f.q[0] = *(const u32x4*)(r + hi * 8);
  f.q[1] = *(const u32x4*)(r + 16 + hi * 8);
  return f.v;
}

__device__ __forceinline__ v16bf load_frag_b(const bf16* base, int ld, int lane) {
  const int n = lane & 15, hi = lane >> 4;
  const bf16* r = base + n * ld;
  FragU f;
  f.q[0] = *(const u32x4*)(r + hi * 8);
  f.q[1] = *(const u32x4*)(r + 16 + hi * 8);
  return f.v;
}

// ---------------- generic tiled WMMA GEMM:  C = scale*(A @ W^T) + bias -----
// A: [M,K] bf16 row-major (lda), W: [N,K] bf16 row-major (ldw).
// Two-level batch: z -> (zo = z/zInner, zi = z%zInner); base += zo*so + zi*si.
// Block = 256 thr = 8 waves; C tile 128x128; wave = 32(M) x 64(N) = 8 WMMAs.
// Double-buffered async global->LDS staging pipeline.
template <typename OutT>
__global__ __launch_bounds__(256) void gemm_bf16_wmma(
    const bf16* __restrict__ A, long sAo, long sAi, int lda,
    const bf16* __restrict__ W, long sWo, long sWi, int ldw,
    const float* __restrict__ bias,
    OutT* __restrict__ C, long sCo, long sCi, int ldc,
    int zInner, int M, int N, int K, float scale)
{
  __shared__ bf16 lds_a[2][128 * 32];
  __shared__ bf16 lds_w[2][128 * 32];

  const int z  = blockIdx.z;
  const int zo = z / zInner, zi = z % zInner;
  A += (long)zo * sAo + (long)zi * sAi;
  W += (long)zo * sWo + (long)zi * sWi;
  C += (long)zo * sCo + (long)zi * sCi;

  const int bm   = blockIdx.y * 128;
  const int bn   = blockIdx.x * 128;
  const int tid  = threadIdx.x;
  const int lane = tid & 31;
  const int wid  = tid >> 5;
  const int wm   = wid & 3;    // 32-row slab
  const int wn   = wid >> 2;   // 64-col slab

  v8f acc[2][4];
  for (int i = 0; i < 2; ++i)
    for (int j = 0; j < 4; ++j)
      for (int e = 0; e < 8; ++e) acc[i][j][e] = 0.f;

  const int row  = tid >> 1;   // 0..127
  const int half = tid & 1;    // 16-element half of the 32-wide K slice

  int wrow = bn + row; if (wrow >= N) wrow = N - 1;   // clamp for N < 128
  const bf16* srcA = A + (long)(bm + row) * lda + half * 16;
  const bf16* srcW = W + (long)wrow * ldw + half * 16;

  const unsigned offA0 = lds_addr(&lds_a[0][row * 32 + half * 16]);
  const unsigned offW0 = lds_addr(&lds_w[0][row * 32 + half * 16]);
  const unsigned bufStride = 128u * 32u * 2u;   // bytes between double buffers

  // stage tile (k0) into buffer `buf`: 4 async b128 per thread
  auto stage = [&](int buf, int k0) {
    const unsigned oa = offA0 + (unsigned)buf * bufStride;
    const unsigned ow = offW0 + (unsigned)buf * bufStride;
    async_copy_b128(oa,      srcA + k0);
    async_copy_b128(oa + 16, srcA + k0 + 8);
    async_copy_b128(ow,      srcW + k0);
    async_copy_b128(ow + 16, srcW + k0 + 8);
  };

  const int nt = K / 32;
  stage(0, 0);

  for (int kk = 0; kk < nt; ++kk) {
    const int cur = kk & 1;
    const bool more = (kk + 1) < nt;
    __syncthreads();                       // all waves done reading buf[cur^1]
    if (more) stage(cur ^ 1, (kk + 1) * 32);
    if (more) wait_async_le4();            // tile kk's 4 copies have landed
    else      wait_async_0();
    __syncthreads();                       // publish buf[cur] to all waves

    const bf16* abase = &lds_a[cur][(wm * 32) * 32];
    const bf16* wbase = &lds_w[cur][(wn * 64) * 32];
    v16bf af[2], bfr[4];
    for (int i = 0; i < 2; ++i) af[i]  = load_frag_a(abase + (i * 16) * 32, 32, lane);
    for (int j = 0; j < 4; ++j) bfr[j] = load_frag_b(wbase + (j * 16) * 32, 32, lane);
    for (int i = 0; i < 2; ++i)
      for (int j = 0; j < 4; ++j)
        acc[i][j] = __builtin_amdgcn_wmma_f32_16x16x32_bf16(
            false, af[i], false, bfr[j], (short)0, acc[i][j], false, false);
  }

  // C layout: VGPR r -> M = r + 8*(lane>>4); N = lane&15.
  const int nlane = lane & 15;
  const int moff  = (lane >> 4) * 8;
  for (int i = 0; i < 2; ++i) {
    const int m0 = bm + wm * 32 + i * 16 + moff;
    for (int j = 0; j < 4; ++j) {
      const int n = bn + wn * 64 + j * 16 + nlane;
      if (n >= N) continue;
      const float bv = bias ? bias[n] : 0.f;
      for (int r = 0; r < 8; ++r) {
        const int m = m0 + r;
        if (m < M) C[(long)m * ldc + n] = (OutT)(acc[i][j][r] * scale + bv);
      }
    }
  }
}

// ---------------- helpers ---------------------------------------------------
__device__ __forceinline__ float block_reduce_sum(float v, float* sbuf) {
  const int tid = threadIdx.x;
  sbuf[tid] = v; __syncthreads();
  for (int s = 128; s > 0; s >>= 1) {
    if (tid < s) sbuf[tid] += sbuf[tid + s];
    __syncthreads();
  }
  const float r = sbuf[0]; __syncthreads();
  return r;
}

// masks + fp32 -> bf16 conversion of activations (one row per block)
__global__ __launch_bounds__(256) void prep_rows(
    const float* __restrict__ tgt, const float* __restrict__ s1, const float* __restrict__ s2,
    bf16* __restrict__ tgtb, bf16* __restrict__ s1b, bf16* __restrict__ s2b,
    float* __restrict__ tmask, float* __restrict__ m1, float* __restrict__ m2)
{
  __shared__ float sbuf[256];
  const int r = blockIdx.x;
  const float* src; bf16* dst; float* msk; int lrow;
  if (r < B_ * T_)             { src = tgt; dst = tgtb; msk = tmask; lrow = r; }
  else if (r < B_ * (T_ + S_)) { src = s1;  dst = s1b;  msk = m1;    lrow = r - B_ * T_; }
  else                         { src = s2;  dst = s2b;  msk = m2;    lrow = r - B_ * (T_ + S_); }
  const float* x = src + (long)lrow * D_;
  bf16* y = dst + (long)lrow * D_;
  float a = 0.f;
  for (int d = threadIdx.x; d < D_; d += 256) {
    const float v = x[d]; a += fabsf(v); y[d] = (bf16)v;
  }
  const float s = block_reduce_sum(a, sbuf);
  if (threadIdx.x == 0) msk[lrow] = (s != 0.f) ? 1.f : 0.f;
}

__global__ void f32_to_bf16(const float* __restrict__ s, bf16* __restrict__ d, long n) {
  const long i = (long)blockIdx.x * blockDim.x + threadIdx.x;
  if (i < n) d[i] = (bf16)s[i];
}

// v [B,S,H*DH] -> vt [B,H,DH,S]
__global__ void transpose_v(const bf16* __restrict__ v, bf16* __restrict__ vt) {
  const long i = (long)blockIdx.x * blockDim.x + threadIdx.x;
  if (i >= (long)B_ * H_ * DH_ * S_) return;
  const int s = (int)(i % S_); const long t1 = i / S_;
  const int d = (int)(t1 % DH_); const long t2 = t1 / DH_;
  const int h = (int)(t2 % H_); const int b = (int)(t2 / H_);
  vt[i] = v[((long)(b * S_ + s)) * HID_ + h * DH_ + d];
}

// mean over heads (with masks) + layer-norm over S -> ln_att [B,T,S]
__global__ __launch_bounds__(256) void attmean_ln(
    const float* __restrict__ logits, const float* __restrict__ qmask,
    const float* __restrict__ kmask, float* __restrict__ out)
{
  __shared__ float sbuf[256];
  const int bt = blockIdx.x;
  const int b = bt / T_, t = bt % T_;
  const int s = threadIdx.x;
  float acc = 0.f;
  for (int h = 0; h < H_; ++h)
    acc += logits[((long)(b * H_ + h) * T_ + t) * S_ + s];
  const float val  = acc * (1.f / H_) * kmask[b * S_ + s] * qmask[bt];
  const float mean = block_reduce_sum(val, sbuf) * (1.f / S_);
  const float dv   = val - mean;
  const float var  = block_reduce_sum(dv * dv, sbuf) * (1.f / S_);
  out[(long)bt * S_ + s] = dv * rsqrtf(var + 1e-5f);
}

// softmax over S with kmask==0 -> -inf, write bf16 probabilities
__global__ __launch_bounds__(256) void softmax_rows(
    const float* __restrict__ logits, const float* __restrict__ kmask,
    bf16* __restrict__ sm)
{
  __shared__ float sbuf[256];
  const long row = blockIdx.x;                 // (b*H + h)*T + t
  const int b = (int)(row / (H_ * T_));
  const int s = threadIdx.x;
  const float km = kmask[b * S_ + s];
  const float x  = (km == 0.f) ? -1e30f : logits[row * S_ + s];
  sbuf[s] = x; __syncthreads();
  for (int st = 128; st > 0; st >>= 1) {
    if (s < st) sbuf[s] = fmaxf(sbuf[s], sbuf[s + st]);
    __syncthreads();
  }
  const float mx = sbuf[0]; __syncthreads();
  const float e   = __expf(x - mx);
  const float sum = block_reduce_sum(e, sbuf);
  sm[row * S_ + s] = (bf16)(e / sum);
}

// gate p = softmax([tgt, c1*qm, c2*qm] @ Wp.T + bp)  -> [B*T, 3]
__global__ __launch_bounds__(256) void gate_p(
    const float* __restrict__ tgt, const float* __restrict__ c1, const float* __restrict__ c2,
    const float* __restrict__ qmask, const float* __restrict__ Wp, const float* __restrict__ bp,
    float* __restrict__ p)
{
  __shared__ float sbuf[256];
  const int row = blockIdx.x;
  const float qm = qmask[row];
  float part[3] = {0.f, 0.f, 0.f};
  for (int d = threadIdx.x; d < 3 * D_; d += 256) {
    float x;
    if (d < D_)          x = tgt[(long)row * D_ + d];
    else if (d < 2 * D_) x = c1[(long)row * D_ + (d - D_)] * qm;
    else                 x = c2[(long)row * D_ + (d - 2 * D_)] * qm;
    for (int j = 0; j < 3; ++j) part[j] += x * Wp[j * (3 * D_) + d];
  }
  float dot[3];
  for (int j = 0; j < 3; ++j) dot[j] = block_reduce_sum(part[j], sbuf);
  if (threadIdx.x == 0) {
    const float l0 = dot[0] + bp[0], l1 = dot[1] + bp[1], l2 = dot[2] + bp[2];
    const float mx = fmaxf(l0, fmaxf(l1, l2));
    const float e0 = __expf(l0 - mx), e1 = __expf(l1 - mx), e2 = __expf(l2 - mx);
    const float inv = 1.f / (e0 + e1 + e2);
    p[row * 3 + 0] = e0 * inv; p[row * 3 + 1] = e1 * inv; p[row * 3 + 2] = e2 * inv;
  }
}

// in-place layer-norm over [0,V) of each output row, scale by p0, zero pad
__global__ __launch_bounds__(256) void finalize_ln(
    float* __restrict__ out, const float* __restrict__ p)
{
  __shared__ float sbuf[256];
  const int row = blockIdx.x;
  float* x = out + (long)row * VEXT_;
  float s = 0.f, s2 = 0.f;
  for (int i = threadIdx.x; i < V_; i += 256) { const float v = x[i]; s += v; s2 += v * v; }
  const float mean = block_reduce_sum(s,  sbuf) * (1.f / V_);
  const float msq  = block_reduce_sum(s2, sbuf) * (1.f / V_);
  const float rinv = rsqrtf(fmaxf(msq - mean * mean, 0.f) + 1e-5f);
  const float p0 = p[row * 3];
  for (int i = threadIdx.x; i < V_; i += 256) x[i] = (x[i] - mean) * rinv * p0;
  for (int i = V_ + threadIdx.x; i < VEXT_; i += 256) x[i] = 0.f;
}

// out[b,t,map[b,s]] += p[b,t,pcol] * ln_att[b,t,s]
__global__ void scatter_add(const float* __restrict__ ln_att, const int* __restrict__ map,
                            const float* __restrict__ p, int pcol, float* __restrict__ out)
{
  const long i = (long)blockIdx.x * blockDim.x + threadIdx.x;
  if (i >= (long)B_ * T_ * S_) return;
  const int s  = (int)(i % S_);
  const long bt = i / S_;
  const int b  = (int)(bt / T_);
  const int idx = map[b * S_ + s];
  const float val = p[bt * 3 + pcol] * ln_att[i];
  atomicAdd(out + bt * VEXT_ + idx, val);
}

// ---------------------------------------------------------------------------
extern "C" void kernel_launch(void* const* d_in, const int* in_sizes, int n_in,
                              void* d_out, int out_size, void* d_ws, size_t ws_size,
                              hipStream_t stream) {
  (void)in_sizes; (void)n_in; (void)out_size; (void)ws_size;

  const float* tgt  = (const float*)d_in[0];
  const float* s1k  = (const float*)d_in[1];
  const int*   map1 = (const int*)  d_in[2];
  const float* s2k  = (const float*)d_in[3];
  const int*   map2 = (const int*)  d_in[4];
  const float* Wfc  = (const float*)d_in[5];
  const float* bfc  = (const float*)d_in[6];
  const float* Wp   = (const float*)d_in[7];
  const float* bp   = (const float*)d_in[8];
  const float* Wf32[8]; const float* bias[8];
  for (int i = 0; i < 8; ++i) { Wf32[i] = (const float*)d_in[9 + 2 * i]; bias[i] = (const float*)d_in[10 + 2 * i]; }
  // order: Wq1 Wk1 Wv1 Wo1 Wq2 Wk2 Wv2 Wo2

  float* outF = (float*)d_out;

  // ---- workspace carve (~82 MB) ----
  char* w = (char*)d_ws;
  auto alloc = [&](size_t bytes) { char* q = w; w += (bytes + 255) & ~(size_t)255; return q; };
  bf16* tgtb  = (bf16*)alloc((size_t)B_ * T_ * D_ * 2);
  bf16* s1b   = (bf16*)alloc((size_t)B_ * S_ * D_ * 2);
  bf16* s2b   = (bf16*)alloc((size_t)B_ * S_ * D_ * 2);
  float* tmask = (float*)alloc((size_t)B_ * T_ * 4);
  float* m1    = (float*)alloc((size_t)B_ * S_ * 4);
  float* m2    = (float*)alloc((size_t)B_ * S_ * 4);
  bf16* Wfcb  = (bf16*)alloc((size_t)V_ * D_ * 2);
  bf16* Wb[8];
  for (int i = 0; i < 8; ++i) Wb[i] = (bf16*)alloc((size_t)HID_ * D_ * 2);
  bf16* q1 = (bf16*)alloc((size_t)B_ * T_ * HID_ * 2);
  bf16* k1 = (bf16*)alloc((size_t)B_ * S_ * HID_ * 2);
  bf16* v1 = (bf16*)alloc((size_t)B_ * S_ * HID_ * 2);
  bf16* q2 = (bf16*)alloc((size_t)B_ * T_ * HID_ * 2);
  bf16* k2 = (bf16*)alloc((size_t)B_ * S_ * HID_ * 2);
  bf16* v2 = (bf16*)alloc((size_t)B_ * S_ * HID_ * 2);
  bf16* vt1 = (bf16*)alloc((size_t)B_ * H_ * DH_ * S_ * 2);
  bf16* vt2 = (bf16*)alloc((size_t)B_ * H_ * DH_ * S_ * 2);
  float* log1 = (float*)alloc((size_t)B_ * H_ * T_ * S_ * 4);
  float* log2 = (float*)alloc((size_t)B_ * H_ * T_ * S_ * 4);
  bf16* sm1  = (bf16*)alloc((size_t)B_ * H_ * T_ * S_ * 2);
  bf16* sm2  = (bf16*)alloc((size_t)B_ * H_ * T_ * S_ * 2);
  float* lnat1 = (float*)alloc((size_t)B_ * T_ * S_ * 4);
  float* lnat2 = (float*)alloc((size_t)B_ * T_ * S_ * 4);
  bf16* ctx1 = (bf16*)alloc((size_t)B_ * T_ * HID_ * 2);
  bf16* ctx2 = (bf16*)alloc((size_t)B_ * T_ * HID_ * 2);
  float* c1f = (float*)alloc((size_t)B_ * T_ * D_ * 4);
  float* c2f = (float*)alloc((size_t)B_ * T_ * D_ * 4);
  float* pg  = (float*)alloc((size_t)B_ * T_ * 3 * 4);

  const int MT = B_ * T_;   // 1024

  // 1) masks + activation conversion
  prep_rows<<<B_ * (T_ + S_ + S_), 256, 0, stream>>>(tgt, s1k, s2k, tgtb, s1b, s2b, tmask, m1, m2);

  // 2) weight conversions
  {
    const long nfc = (long)V_ * D_;
    f32_to_bf16<<<(unsigned)((nfc + 255) / 256), 256, 0, stream>>>(Wfc, Wfcb, nfc);
    const long nw = (long)HID_ * D_;
    for (int i = 0; i < 8; ++i)
      f32_to_bf16<<<(unsigned)((nw + 255) / 256), 256, 0, stream>>>(Wf32[i], Wb[i], nw);
  }

  // 3) fc GEMM -> d_out[:, :V]  (M=1024, N=32000, K=512)
  gemm_bf16_wmma<float><<<dim3(V_ / 128, MT / 128, 1), 256, 0, stream>>>(
      tgtb, 0, 0, D_, Wfcb, 0, 0, D_, bfc, outF, 0, 0, VEXT_, 1, MT, V_, D_, 1.f);

  // 4) Q/K/V projections (bf16 out)
  gemm_bf16_wmma<bf16><<<dim3(HID_ / 128, MT / 128, 1), 256, 0, stream>>>(
      tgtb, 0, 0, D_, Wb[0], 0, 0, D_, bias[0], q1, 0, 0, HID_, 1, MT, HID_, D_, 1.f);
  gemm_bf16_wmma<bf16><<<dim3(HID_ / 128, MT / 128, 1), 256, 0, stream>>>(
      s1b, 0, 0, D_, Wb[1], 0, 0, D_, bias[1], k1, 0, 0, HID_, 1, MT, HID_, D_, 1.f);
  gemm_bf16_wmma<bf16><<<dim3(HID_ / 128, MT / 128, 1), 256, 0, stream>>>(
      s1b, 0, 0, D_, Wb[2], 0, 0, D_, bias[2], v1, 0, 0, HID_, 1, MT, HID_, D_, 1.f);
  gemm_bf16_wmma<bf16><<<dim3(HID_ / 128, MT / 128, 1), 256, 0, stream>>>(
      tgtb, 0, 0, D_, Wb[4], 0, 0, D_, bias[4], q2, 0, 0, HID_, 1, MT, HID_, D_, 1.f);
  gemm_bf16_wmma<bf16><<<dim3(HID_ / 128, MT / 128, 1), 256, 0, stream>>>(
      s2b, 0, 0, D_, Wb[5], 0, 0, D_, bias[5], k2, 0, 0, HID_, 1, MT, HID_, D_, 1.f);
  gemm_bf16_wmma<bf16><<<dim3(HID_ / 128, MT / 128, 1), 256, 0, stream>>>(
      s2b, 0, 0, D_, Wb[6], 0, 0, D_, bias[6], v2, 0, 0, HID_, 1, MT, HID_, D_, 1.f);

  // 5) transpose V -> [B,H,DH,S]
  {
    const long nv = (long)B_ * H_ * DH_ * S_;
    transpose_v<<<(unsigned)((nv + 255) / 256), 256, 0, stream>>>(v1, vt1);
    transpose_v<<<(unsigned)((nv + 255) / 256), 256, 0, stream>>>(v2, vt2);
  }

  // 6) QK^T batched over (b,h): M=T, N=S, K=DH, scale = DH^-0.5 = 0.125
  gemm_bf16_wmma<float><<<dim3(S_ / 128, T_ / 128, B_ * H_), 256, 0, stream>>>(
      q1, (long)T_ * HID_, DH_, HID_, k1, (long)S_ * HID_, DH_, HID_, nullptr,
      log1, (long)H_ * T_ * S_, (long)T_ * S_, S_, H_, T_, S_, DH_, 0.125f);
  gemm_bf16_wmma<float><<<dim3(S_ / 128, T_ / 128, B_ * H_), 256, 0, stream>>>(
      q2, (long)T_ * HID_, DH_, HID_, k2, (long)S_ * HID_, DH_, HID_, nullptr,
      log2, (long)H_ * T_ * S_, (long)T_ * S_, S_, H_, T_, S_, DH_, 0.125f);

  // 7) head-mean + layer-norm of masked logits
  attmean_ln<<<MT, 256, 0, stream>>>(log1, tmask, m1, lnat1);
  attmean_ln<<<MT, 256, 0, stream>>>(log2, tmask, m2, lnat2);

  // 8) softmax rows -> bf16
  softmax_rows<<<B_ * H_ * T_, 256, 0, stream>>>(log1, m1, sm1);
  softmax_rows<<<B_ * H_ * T_, 256, 0, stream>>>(log2, m2, sm2);

  // 9) context = softmax @ V : M=T, N=DH(64), K=S, batched over (b,h)
  gemm_bf16_wmma<bf16><<<dim3(1, T_ / 128, B_ * H_), 256, 0, stream>>>(
      sm1, (long)H_ * T_ * S_, (long)T_ * S_, S_, vt1, (long)H_ * DH_ * S_, (long)DH_ * S_, S_,
      nullptr, ctx1, (long)T_ * HID_, DH_, HID_, H_, T_, DH_, S_, 1.f);
  gemm_bf16_wmma<bf16><<<dim3(1, T_ / 128, B_ * H_), 256, 0, stream>>>(
      sm2, (long)H_ * T_ * S_, (long)T_ * S_, S_, vt2, (long)H_ * DH_ * S_, (long)DH_ * S_, S_,
      nullptr, ctx2, (long)T_ * HID_, DH_, HID_, H_, T_, DH_, S_, 1.f);

  // 10) output projections c = ctx @ Wo^T + bo
  gemm_bf16_wmma<float><<<dim3(D_ / 128, MT / 128, 1), 256, 0, stream>>>(
      ctx1, 0, 0, HID_, Wb[3], 0, 0, HID_, bias[3], c1f, 0, 0, D_, 1, MT, D_, HID_, 1.f);
  gemm_bf16_wmma<float><<<dim3(D_ / 128, MT / 128, 1), 256, 0, stream>>>(
      ctx2, 0, 0, HID_, Wb[7], 0, 0, HID_, bias[7], c2f, 0, 0, D_, 1, MT, D_, HID_, 1.f);

  // 11) gate probabilities
  gate_p<<<MT, 256, 0, stream>>>(tgt, c1f, c2f, tmask, Wp, bp, pg);

  // 12) layer-norm fc rows in place, scale by p0, zero the pad region
  finalize_ln<<<MT, 256, 0, stream>>>(outF, pg);

  // 13) fused scatter-copy of both sources
  {
    const long ns = (long)B_ * T_ * S_;
    scatter_add<<<(unsigned)((ns + 255) / 256), 256, 0, stream>>>(lnat1, map1, pg, 1, outF);
    scatter_add<<<(unsigned)((ns + 255) / 256), 256, 0, stream>>>(lnat2, map2, pg, 2, outF);
  }
}
